// Waveletnet_46145128628487
// MI455X (gfx1250) — matrix operand order repaired and checked by reading, not compiled, not run on hardware
//
#include <hip/hip_runtime.h>

// ---------------------------------------------------------------------------
// CDNA5 (gfx1250) wavelet U-Net, NHWC-f16 activations, implicit-GEMM convs on
// v_wmma_f32_16x16x32_f16 (f16 in, f32 accumulate). wave32 only.
// GEMM K-order: k = (kh*3+kw)*Cin + ci  -> contiguous b128 loads everywhere.
// Conv inputs use a spatially padded layout (N, H+2, W+2, C) with a zeroed
// 1-pixel border ring: 3x3 taps need no bounds checks and no cndmasks.
// ---------------------------------------------------------------------------

typedef __attribute__((ext_vector_type(16))) _Float16 v16h;
typedef __attribute__((ext_vector_type(8)))  _Float16 h8;
typedef __attribute__((ext_vector_type(8)))  float    v8f;
typedef __attribute__((ext_vector_type(4)))  float    f4;

__device__ __forceinline__ float lrelu_f(float v) { return v > 0.0f ? v : 0.2f * v; }

// ---------------------------------------------------------------------------
// Weight standardization. Input w: OIHW f32, (Cout, CinReal*9) per filter.
// Output: f16, [Cout][Kpad], k = (kh*3+kw)*CinPad + ci, zero padded for
// ci >= CinReal and k >= 9*CinPad. Stats over the real fan-in, ddof=1.
// ---------------------------------------------------------------------------
__global__ void k_wstd_f16(const float* __restrict__ w, _Float16* __restrict__ out,
                           int CinReal, int CinPad, int Kpad) {
    __shared__ float red[256];
    __shared__ float s_mean, s_inv;
    const int fanin = CinReal * 9;
    const int o = blockIdx.x;
    const float* wr = w + (size_t)o * fanin;

    float s = 0.0f;
    for (int i = threadIdx.x; i < fanin; i += blockDim.x) s += wr[i];
    red[threadIdx.x] = s;
    __syncthreads();
    for (int st = 128; st > 0; st >>= 1) {
        if (threadIdx.x < st) red[threadIdx.x] += red[threadIdx.x + st];
        __syncthreads();
    }
    if (threadIdx.x == 0) s_mean = red[0] / (float)fanin;
    __syncthreads();
    const float mean = s_mean;

    float ss = 0.0f;
    for (int i = threadIdx.x; i < fanin; i += blockDim.x) {
        float c = wr[i] - mean;
        ss += c * c;
    }
    red[threadIdx.x] = ss;
    __syncthreads();
    for (int st = 128; st > 0; st >>= 1) {
        if (threadIdx.x < st) red[threadIdx.x] += red[threadIdx.x + st];
        __syncthreads();
    }
    if (threadIdx.x == 0)
        s_inv = 1.0f / (sqrtf(red[0] / (float)(fanin - 1)) + 1e-5f);
    __syncthreads();
    const float inv = s_inv;

    for (int j = threadIdx.x; j < Kpad; j += blockDim.x) {
        int sp = j / CinPad;          // spatial slice kh*3+kw
        int ci = j - sp * CinPad;
        _Float16 v = (_Float16)0.0f;
        if (sp < 9 && ci < CinReal)
            v = (_Float16)((wr[ci * 9 + sp] - mean) * inv);
        out[(size_t)o * Kpad + j] = v;
    }
}

// Tiny f32 standardization for the final 1x1 conv (2 filters x 3 inputs).
__global__ void k_wstd_final(const float* __restrict__ w, float* __restrict__ out) {
    int o = threadIdx.x;
    if (o < 2) {
        float a = w[o * 3 + 0], b = w[o * 3 + 1], c = w[o * 3 + 2];
        float m = (a + b + c) * (1.0f / 3.0f);
        float ca = a - m, cb = b - m, cc = c - m;
        float inv = 1.0f / (sqrtf((ca * ca + cb * cb + cc * cc) * 0.5f) + 1e-5f);
        out[o * 3 + 0] = ca * inv;
        out[o * 3 + 1] = cb * inv;
        out[o * 3 + 2] = cc * inv;
    }
}

// Zero the 1-pixel border ring of a padded NHWC f16 tensor (N, H+2, W+2, C).
__global__ void k_zero_border(_Float16* __restrict__ t, int N, int H, int W, int C) {
    const int Hp = H + 2, Wp = W + 2;
    const int ring = 2 * Wp + 2 * H;
    const long total = (long)N * ring * C;
    long idx = (long)blockIdx.x * blockDim.x + threadIdx.x;
    if (idx >= total) return;
    int c = idx % C;
    long u = idx / C;
    int r = u % ring;
    int n = u / ring;
    int prow, pcol;
    if (r < Wp) { prow = 0; pcol = r; }
    else if (r < 2 * Wp) { prow = Hp - 1; pcol = r - Wp; }
    else {
        int rr = r - 2 * Wp;
        prow = 1 + (rr >> 1);
        pcol = (rr & 1) ? (Wp - 1) : 0;
    }
    t[(((size_t)n * Hp + prow) * Wp + pcol) * C + c] = (_Float16)0.0f;
}

// ---------------------------------------------------------------------------
// Haar wt from the network input: x NCHW f32 (N,3,H,W) ->
// padded NHWC f16 (N,H/2+2,W/2+2,16), channels c*4+band, ch 12..15 zero.
// ---------------------------------------------------------------------------
__global__ void k_wt_x(const float* __restrict__ x, _Float16* __restrict__ out,
                       int N, int H, int W) {
    const int H2 = H >> 1, W2 = W >> 1;
    const long total = (long)N * H2 * W2;
    long idx = (long)blockIdx.x * blockDim.x + threadIdx.x;
    if (idx >= total) return;
    int ow = idx % W2;
    long t = idx / W2;
    int oh = t % H2;
    int n = t / H2;
    _Float16* q = out + (((size_t)n * (H2 + 2) + oh + 1) * (W2 + 2) + ow + 1) * 16;
#pragma unroll
    for (int c = 0; c < 3; ++c) {
        const float* p = x + (((size_t)n * 3 + c) * H + 2 * oh) * W + 2 * ow;
        float a = p[0], b = p[1], cc = p[W], d = p[W + 1];
        q[c * 4 + 0] = (_Float16)(0.25f * (a + b + cc + d));
        q[c * 4 + 1] = (_Float16)((0.5f * (a + b - cc - d) + 1.0f) * 0.5f);
        q[c * 4 + 2] = (_Float16)((0.5f * (a - b + cc - d) + 1.0f) * 0.5f);
        q[c * 4 + 3] = (_Float16)((0.5f * (a - b - cc + d) + 1.0f) * 0.5f);
    }
#pragma unroll
    for (int c = 12; c < 16; ++c) q[c] = (_Float16)0.0f;
}

// Haar wt: dense NHWC f16 (N,H,W,C) -> padded NHWC f16 (N,H/2+2,W/2+2,4C).
__global__ void k_wt_h(const _Float16* __restrict__ in, _Float16* __restrict__ out,
                       int N, int C, int H, int W) {
    const int H2 = H >> 1, W2 = W >> 1;
    const long total = (long)N * H2 * W2 * C;
    long idx = (long)blockIdx.x * blockDim.x + threadIdx.x;
    if (idx >= total) return;
    int c = idx % C;
    long t = idx / C;
    int ow = t % W2; t /= W2;
    int oh = t % H2;
    int n = t / H2;
    const _Float16* p = in + (((size_t)n * H + 2 * oh) * W + 2 * ow) * C + c;
    float a = (float)p[0];
    float b = (float)p[C];
    float cc = (float)p[(size_t)W * C];
    float d = (float)p[(size_t)W * C + C];
    _Float16* q = out +
        (((size_t)n * (H2 + 2) + oh + 1) * (W2 + 2) + ow + 1) * (4 * C) + 4 * c;
    q[0] = (_Float16)(0.25f * (a + b + cc + d));
    q[1] = (_Float16)((0.5f * (a + b - cc - d) + 1.0f) * 0.5f);
    q[2] = (_Float16)((0.5f * (a - b + cc - d) + 1.0f) * 0.5f);
    q[3] = (_Float16)((0.5f * (a - b - cc + d) + 1.0f) * 0.5f);
}

// ---------------------------------------------------------------------------
// Inverse Haar wt, dense NHWC f16 in -> NHWC f16 out (dense or padded),
// written at channel offset chanOff (fused concat). Bands un-remapped (2v-1).
// ---------------------------------------------------------------------------
__global__ void k_iwt_h(const _Float16* __restrict__ v, _Float16* __restrict__ dst,
                        int N, int C, int Hs, int Ws, int Cdst, int chanOff,
                        int outPadded) {
    const int OH = Hs * 2, OW = Ws * 2;
    const long total = (long)N * OH * OW * C;
    long idx = (long)blockIdx.x * blockDim.x + threadIdx.x;
    if (idx >= total) return;
    int c = idx % C;
    long t = idx / C;
    int ox = t % OW; t /= OW;
    int oy = t % OH;
    int n = t / OH;
    int y = oy >> 1, i = oy & 1;
    int x = ox >> 1, j = ox & 1;
    const _Float16* p = v + (((size_t)n * Hs + y) * Ws + x) * (4 * C) + 4 * c;
    float r0 = (float)p[0];
    float r1 = 2.0f * (float)p[1] - 1.0f;
    float r2 = 2.0f * (float)p[2] - 1.0f;
    float r3 = 2.0f * (float)p[3] - 1.0f;
    float s1 = i ? -1.0f : 1.0f;
    float s2 = j ? -1.0f : 1.0f;
    float o = r0 + 0.5f * (s1 * r1 + s2 * r2 + s1 * s2 * r3);
    size_t di;
    if (outPadded)
        di = (((size_t)n * (OH + 2) + oy + 1) * (OW + 2) + ox + 1) * Cdst + chanOff + c;
    else
        di = (((size_t)n * OH + oy) * OW + ox) * Cdst + chanOff + c;
    dst[di] = (_Float16)o;
}

// Dense NHWC (N,H,W,Cs) -> padded NHWC (N,H+2,W+2,Cdst) at chanOff.
// H = 1<<(shHW-shW), W = 1<<shW.
__global__ void k_copy_ch_pad(const _Float16* __restrict__ src, _Float16* __restrict__ dst,
                              long NP, int Cs, int Cdst, int chanOff,
                              int shW, int shHW) {
    const long total = NP * Cs;
    long idx = (long)blockIdx.x * blockDim.x + threadIdx.x;
    if (idx >= total) return;
    int cs = idx % Cs;
    long p = idx / Cs;
    int n = (int)(p >> shHW);
    int hw = (int)(p & ((1 << shHW) - 1));
    int oh = hw >> shW;
    int ow = hw & ((1 << shW) - 1);
    int W = 1 << shW;
    int H = 1 << (shHW - shW);
    dst[(((size_t)n * (H + 2) + oh + 1) * (W + 2) + ow + 1) * Cdst + chanOff + cs] =
        src[idx];
}

// ---------------------------------------------------------------------------
// Implicit-GEMM 3x3 conv (stride 1, pad 1) via WMMA f16->f32.
// Input: padded NHWC f16 (N,H+2,W+2,Cin), zeroed border -> no bounds checks.
// One wave per 32(M) x 64(N) tile: 8 WMMAs per 32-wide K chunk.
// Output: dense NHWC (f32 or f16). Optional bias / padded-f16 skip / lrelu.
// ---------------------------------------------------------------------------
__global__ void k_conv3x3_wmma(const _Float16* __restrict__ in,
                               const _Float16* __restrict__ wstd,
                               const float* __restrict__ bias,
                               const _Float16* __restrict__ skip,
                               void* __restrict__ outv,
                               int N, int Cin, int Cout, int H, int W,
                               int Kpad, int applyLrelu, int outHalf) {
    const int Wp = W + 2, Hp = H + 2;
    const int lane = threadIdx.x;
    const int laneLo = lane & 15;
    const int laneHi = lane >> 4;
    const int Ktot = Cin * 9;
    const int mbase = blockIdx.y * 32;
    const int mA0 = min(mbase + laneLo, Cout - 1);
    const int mA1 = min(mbase + 16 + laneLo, Cout - 1);
    const _Float16* wrow0 = wstd + (size_t)mA0 * Kpad;
    const _Float16* wrow1 = wstd + (size_t)mA1 * Kpad;

    int ohs[4], nis[4], owl[4];
    const _Float16* pbase[4];
#pragma unroll
    for (int j = 0; j < 4; ++j) {
        int pos = (blockIdx.x * 4 + j) * 16;
        int ow0 = pos % W;
        int t = pos / W;
        ohs[j] = t % H;
        nis[j] = t / H;
        owl[j] = ow0 + laneLo;
        // padded-space origin of the (kh=0,kw=0) tap for this lane's column
        pbase[j] = in + (((size_t)nis[j] * Hp + ohs[j]) * Wp + owl[j]) * Cin;
    }

    v8f acc[2][4];
    v8f zf = {};
#pragma unroll
    for (int mt = 0; mt < 2; ++mt)
#pragma unroll
        for (int j = 0; j < 4; ++j) acc[mt][j] = zf;

    // Incremental (slice, ci-base) tracking for this lane-half.
    int cib = laneHi * 16;
    int sl = 0;
    if (cib >= Cin) { cib -= Cin; sl = 1; }   // Cin >= 16 -> at most once

    for (int kb = 0; kb < Ktot; kb += 32) {
        // Two A fragments (rows mbase.. and mbase+16..), contiguous runs.
        h8 a00 = *(const h8*)(wrow0 + kb + laneHi * 16);
        h8 a01 = *(const h8*)(wrow0 + kb + laneHi * 16 + 8);
        h8 a10 = *(const h8*)(wrow1 + kb + laneHi * 16);
        h8 a11 = *(const h8*)(wrow1 + kb + laneHi * 16 + 8);
        v16h A0, A1;
#pragma unroll
        for (int e = 0; e < 8; ++e) {
            A0[e] = a00[e]; A0[8 + e] = a01[e];
            A1[e] = a10[e]; A1[8 + e] = a11[e];
        }

        // Tail (sl==9, Cin==16 only): clamp to slice 8 -> loads stay in
        // bounds; the zero-padded A rows kill the bogus contribution.
        const int slc = min(sl, 8);
        const int kh = slc / 3;
        const int kw = slc - kh * 3;
        const int offc = (kh * Wp + kw) * Cin + cib;

#pragma unroll
        for (int j = 0; j < 4; ++j) {
            const _Float16* bp = pbase[j] + offc;
            h8 b0 = *(const h8*)bp;
            h8 b1 = *(const h8*)(bp + 8);
            v16h B;
#pragma unroll
            for (int e = 0; e < 8; ++e) { B[e] = b0[e]; B[8 + e] = b1[e]; }
            acc[0][j] = __builtin_amdgcn_wmma_f32_16x16x32_f16(
                false, A0, false, B, (short)0, acc[0][j], false, false);
            acc[1][j] = __builtin_amdgcn_wmma_f32_16x16x32_f16(
                false, A1, false, B, (short)0, acc[1][j], false, false);
        }

        cib += 32;
        if (cib >= Cin) { cib -= Cin; ++sl; }
        if (cib >= Cin) { cib -= Cin; ++sl; }
    }

    // ---------------- Epilogue ----------------
#pragma unroll
    for (int mt = 0; mt < 2; ++mt) {
        const int cbase = mbase + mt * 16 + laneHi * 8;  // 8 consecutive channels
        const bool wide = (mbase + mt * 16 + 16 <= Cout);
        if (!wide && cbase >= Cout) continue;
#pragma unroll
        for (int j = 0; j < 4; ++j) {
            size_t pix = ((size_t)nis[j] * H + ohs[j]) * W + owl[j];
            size_t base = pix * Cout + cbase;
            // skip tensor lives in the padded layout
            size_t skbase =
                (((size_t)nis[j] * Hp + ohs[j] + 1) * Wp + owl[j] + 1) * Cout + cbase;
            float vr[8];
#pragma unroll
            for (int r = 0; r < 8; ++r) vr[r] = acc[mt][j][r];
            if (wide) {
                if (bias) {
#pragma unroll
                    for (int r = 0; r < 8; ++r) vr[r] += bias[cbase + r];
                }
                if (skip) {
                    h8 sk = *(const h8*)(skip + skbase);
#pragma unroll
                    for (int r = 0; r < 8; ++r) vr[r] += (float)sk[r];
                }
                if (applyLrelu) {
#pragma unroll
                    for (int r = 0; r < 8; ++r) vr[r] = lrelu_f(vr[r]);
                }
                if (outHalf) {
                    h8 hv;
#pragma unroll
                    for (int r = 0; r < 8; ++r) hv[r] = (_Float16)vr[r];
                    *(h8*)((_Float16*)outv + base) = hv;
                } else {
                    f4 lo, hi;
#pragma unroll
                    for (int r = 0; r < 4; ++r) { lo[r] = vr[r]; hi[r] = vr[4 + r]; }
                    float* op = (float*)outv + base;
                    *(f4*)op = lo;
                    *(f4*)(op + 4) = hi;
                }
            } else {
#pragma unroll
                for (int r = 0; r < 8; ++r) {
                    int mrow = cbase + r;
                    if (mrow < Cout) {
                        float v = vr[r];
                        if (bias) v += bias[mrow];
                        if (skip) v += (float)skip[skbase + r];
                        if (applyLrelu) v = lrelu_f(v);
                        if (outHalf) ((_Float16*)outv)[pix * Cout + mrow] = (_Float16)v;
                        else         ((float*)outv)[pix * Cout + mrow] = v;
                    }
                }
            }
        }
    }
}

// ---------------------------------------------------------------------------
// GroupNorm over lrelu(x), dense NHWC f32 input. Stats: one block per (n,g).
// ---------------------------------------------------------------------------
__global__ void k_gn_stats(const float* __restrict__ x, float* __restrict__ mean,
                           float* __restrict__ rstd, int C, int HW, int G) {
    __shared__ float rs[256], rq[256];
    const int n = blockIdx.x / G;
    const int g = blockIdx.x % G;
    const int Cg = C / G;
    const float* base = x + (size_t)n * HW * C + g * Cg;

    float s = 0.0f, q = 0.0f;
    for (int p = threadIdx.x; p < HW; p += blockDim.x) {
        const float* pp = base + (size_t)p * C;
        for (int cc = 0; cc < Cg; ++cc) {
            float v = lrelu_f(pp[cc]);
            s += v;
            q += v * v;
        }
    }
    rs[threadIdx.x] = s;
    rq[threadIdx.x] = q;
    __syncthreads();
    for (int st = 128; st > 0; st >>= 1) {
        if (threadIdx.x < st) {
            rs[threadIdx.x] += rs[threadIdx.x + st];
            rq[threadIdx.x] += rq[threadIdx.x + st];
        }
        __syncthreads();
    }
    if (threadIdx.x == 0) {
        float cnt = (float)HW * (float)Cg;
        float m = rs[0] / cnt;
        float var = rq[0] / cnt - m * m;
        mean[blockIdx.x] = m;
        rstd[blockIdx.x] = rsqrtf(var + 1e-5f);
    }
}

// Apply: y = (lrelu(x)-mean)*rstd*gamma + beta, dense f32 NHWC in ->
// f16 NHWC out, dense or padded. C, HW, W are powers of two; Cg == 8.
__global__ void k_gn_apply(const float* __restrict__ x, _Float16* __restrict__ out,
                           const float* __restrict__ mean, const float* __restrict__ rstd,
                           const float* __restrict__ gamma, const float* __restrict__ beta,
                           long total, int Cm1, int shC, int shHW, int shW,
                           int outPadded) {
    long idx = (long)blockIdx.x * blockDim.x + threadIdx.x;
    if (idx >= total) return;
    int c = (int)(idx & Cm1);
    int g = c >> 3;                       // Cg = 8
    int G = (Cm1 + 1) >> 3;
    long p = idx >> shC;
    long n = p >> shHW;
    long sid = n * G + g;
    float v = lrelu_f(x[idx]);
    float r = (v - mean[sid]) * rstd[sid] * gamma[c] + beta[c];
    size_t di;
    if (outPadded) {
        int hw = (int)(p & ((1 << shHW) - 1));
        int oh = hw >> shW;
        int ow = hw & ((1 << shW) - 1);
        int W = 1 << shW;
        int H = 1 << (shHW - shW);
        di = (((size_t)n * (H + 2) + oh + 1) * (W + 2) + ow + 1) * (Cm1 + 1) + c;
    } else {
        di = idx;
    }
    out[di] = (_Float16)r;
}

// Final 1x1 WS-conv: dense NHWC f16 (N,HW,3) -> NCHW f32 (N,2,HW).
__global__ void k_final_conv(const _Float16* __restrict__ in, const float* __restrict__ fw,
                             float* __restrict__ out, int N, int HW) {
    const long total = (long)N * 2 * HW;
    long idx = (long)blockIdx.x * blockDim.x + threadIdx.x;
    if (idx >= total) return;
    int hw = idx % HW;
    long t = idx / HW;
    int o = t % 2;
    int n = t / 2;
    const _Float16* p = in + ((size_t)n * HW + hw) * 3;
    out[idx] = fw[o * 3 + 0] * (float)p[0] + fw[o * 3 + 1] * (float)p[1] +
               fw[o * 3 + 2] * (float)p[2];
}

// ---------------------------------------------------------------------------
// Host orchestration
// ---------------------------------------------------------------------------
static inline int ceil_div(long a, long b) { return (int)((a + b - 1) / b); }

extern "C" void kernel_launch(void* const* d_in, const int* in_sizes, int n_in,
                              void* d_out, int out_size, void* d_ws, size_t ws_size,
                              hipStream_t stream) {
    (void)in_sizes; (void)n_in; (void)out_size; (void)ws_size;

    const float* x        = (const float*)d_in[0];
    const float* conv1_w  = (const float*)d_in[1];
    const float* conv1_b  = (const float*)d_in[2];
    const float* conv2_w  = (const float*)d_in[3];
    const float* conv2_b  = (const float*)d_in[4];
    const float* conv3_w  = (const float*)d_in[5];
    const float* conv3_b  = (const float*)d_in[6];
    const float* conv4_w  = (const float*)d_in[7];
    const float* conv4_b  = (const float*)d_in[8];
    const float* convd1_w = (const float*)d_in[9];
    const float* convd1_b = (const float*)d_in[10];
    const float* convd2_w = (const float*)d_in[11];
    const float* convd2_b = (const float*)d_in[12];
    const float* convd3_w = (const float*)d_in[13];
    const float* convd3_b = (const float*)d_in[14];
    const float* convd4_w = (const float*)d_in[15];
    const float* convd4_b = (const float*)d_in[16];
    const float* final_w  = (const float*)d_in[17];
    const float* gn1_w = (const float*)d_in[18];
    const float* gn1_b = (const float*)d_in[19];
    const float* gn2_w = (const float*)d_in[20];
    const float* gn2_b = (const float*)d_in[21];
    const float* gn3_w = (const float*)d_in[22];
    const float* gn3_b = (const float*)d_in[23];
    const float* gn4_w = (const float*)d_in[24];
    const float* gn4_b = (const float*)d_in[25];
    float* y = (float*)d_out;

    size_t off = 0;
    auto alloc = [&](size_t bytes) -> void* {
        void* p = (char*)d_ws + off;
        off += (bytes + 255) & ~(size_t)255;
        return p;
    };

    const int N = 16;
    // Standardized f16 weights, [Cout][Kpad]
    _Float16* w1s = (_Float16*)alloc(sizeof(_Float16) * 16   * 160);
    _Float16* w2s = (_Float16*)alloc(sizeof(_Float16) * 64   * 576);
    _Float16* w3s = (_Float16*)alloc(sizeof(_Float16) * 256  * 2304);
    _Float16* w4s = (_Float16*)alloc(sizeof(_Float16) * 1024 * 9216);
    _Float16* d1s = (_Float16*)alloc(sizeof(_Float16) * 12   * 160);
    _Float16* d2s = (_Float16*)alloc(sizeof(_Float16) * 16   * 288);
    _Float16* d3s = (_Float16*)alloc(sizeof(_Float16) * 64   * 1152);
    _Float16* d4s = (_Float16*)alloc(sizeof(_Float16) * 256  * 4608);
    float* fws   = (float*)alloc(sizeof(float) * 8);
    float* gmean = (float*)alloc(sizeof(float) * 2048);
    float* grstd = (float*)alloc(sizeof(float) * 2048);

    // Activation slots sized for the largest padded tensors they hold
    // (max single: 18*18*1024*16 = 5.31M halves; concat slab: 34*34*512*16 =
    // 9.47M halves), rounded up generously.
    const size_t EH  = 6 * 1024 * 1024;       // halves
    const size_t EH9 = 10 * 1024 * 1024;      // halves
    float*    T  = (float*)alloc(sizeof(float) * 4194304);  // f32 conv temp (dense)
    _Float16* S2 = (_Float16*)alloc(sizeof(_Float16) * EH);   // w1 / iw1
    _Float16* S3 = (_Float16*)alloc(sizeof(_Float16) * EH);   // c1 (dense)
    _Float16* S4 = (_Float16*)alloc(sizeof(_Float16) * EH);   // w2 -> c4 -> ic4
    _Float16* S5 = (_Float16*)alloc(sizeof(_Float16) * EH);   // c2 (dense)
    _Float16* S6 = (_Float16*)alloc(sizeof(_Float16) * EH);   // w3 -> c5 -> ixw
    _Float16* S7 = (_Float16*)alloc(sizeof(_Float16) * EH);   // c3 -> ic2 (dense)
    _Float16* S8 = (_Float16*)alloc(sizeof(_Float16) * EH);   // w4 -> ic3 -> ic1
    _Float16* S9 = (_Float16*)alloc(sizeof(_Float16) * EH9);  // concat slab (padded)

    // --- Weight standardization (padded, spatial-major K) ---
    k_wstd_f16<<<16,   256, 0, stream>>>(conv1_w,  w1s, 12,   16,   160);
    k_wstd_f16<<<64,   256, 0, stream>>>(conv2_w,  w2s, 64,   64,   576);
    k_wstd_f16<<<256,  256, 0, stream>>>(conv3_w,  w3s, 256,  256,  2304);
    k_wstd_f16<<<1024, 256, 0, stream>>>(conv4_w,  w4s, 1024, 1024, 9216);
    k_wstd_f16<<<12,   256, 0, stream>>>(convd1_w, d1s, 16,   16,   160);
    k_wstd_f16<<<16,   256, 0, stream>>>(convd2_w, d2s, 32,   32,   288);
    k_wstd_f16<<<64,   256, 0, stream>>>(convd3_w, d3s, 128,  128,  1152);
    k_wstd_f16<<<256,  256, 0, stream>>>(convd4_w, d4s, 512,  512,  4608);
    k_wstd_final<<<1, 32, 0, stream>>>(final_w, fws);

    auto zb = [&](_Float16* t, int H, int W, int C) {
        long total = (long)N * (2 * (W + 2) + 2 * H) * C;
        k_zero_border<<<ceil_div(total, 256), 256, 0, stream>>>(t, N, H, W, C);
    };
    auto conv = [&](const _Float16* in, const _Float16* ws, const float* b,
                    const _Float16* skip, void* outp, int Cin, int Cout,
                    int H, int Kpad, int lrelu, int outHalf) {
        long P = (long)N * H * H;
        dim3 grid((unsigned)(P / 64), (unsigned)((Cout + 31) / 32));
        k_conv3x3_wmma<<<grid, 32, 0, stream>>>(in, ws, b, skip, outp,
                                                N, Cin, Cout, H, H, Kpad, lrelu, outHalf);
    };
    auto gnorm = [&](const float* in, _Float16* outp, const float* g, const float* b,
                     int C, int HW, int G, int shC, int shHW, int shW, int outPadded) {
        k_gn_stats<<<N * G, 256, 0, stream>>>(in, gmean, grstd, C, HW, G);
        long total = (long)N * C * HW;
        k_gn_apply<<<ceil_div(total, 256), 256, 0, stream>>>(
            in, outp, gmean, grstd, g, b, total, C - 1, shC, shHW, shW, outPadded);
    };
    auto wth = [&](const _Float16* in, _Float16* outp, int C, int H) {
        long total = (long)N * (H / 2) * (H / 2) * C;
        k_wt_h<<<ceil_div(total, 256), 256, 0, stream>>>(in, outp, N, C, H, H);
    };
    auto iwth = [&](const _Float16* in, _Float16* dst, int C, int Hs, int Cdst,
                    int offc, int outPadded) {
        long total = (long)N * (2 * Hs) * (2 * Hs) * C;
        k_iwt_h<<<ceil_div(total, 256), 256, 0, stream>>>(in, dst, N, C, Hs, Hs,
                                                          Cdst, offc, outPadded);
    };
    auto catcp = [&](const _Float16* src, _Float16* dst, int Cs, int Cdst, int offc,
                     int shW, int shHW) {
        long NP = (long)N << shHW;
        k_copy_ch_pad<<<ceil_div(NP * Cs, 256), 256, 0, stream>>>(
            src, dst, NP, Cs, Cdst, offc, shW, shHW);
    };

    // ---------------- Encoder ----------------
    {
        long total = (long)N * 128 * 128;
        k_wt_x<<<ceil_div(total, 256), 256, 0, stream>>>(x, S2, N, 256, 256);  // w1
    }
    zb(S2, 128, 128, 16);
    conv(S2, w1s, conv1_b, nullptr, T, 16, 16, 128, 160, 0, 0);
    gnorm(T, S3, gn1_w, gn1_b, 16, 16384, 2, 4, 14, 7, 0);            // c1 (dense)
    wth(S3, S4, 16, 128);                                             // w2 (64 @64, padded)
    zb(S4, 64, 64, 64);
    conv(S4, w2s, conv2_b, nullptr, T, 64, 64, 64, 576, 0, 0);
    gnorm(T, S5, gn2_w, gn2_b, 64, 4096, 8, 6, 12, 6, 0);             // c2 (dense)
    wth(S5, S6, 64, 64);                                              // w3 (256 @32, padded)
    zb(S6, 32, 32, 256);
    conv(S6, w3s, conv3_b, nullptr, T, 256, 256, 32, 2304, 0, 0);
    gnorm(T, S7, gn3_w, gn3_b, 256, 1024, 32, 8, 10, 5, 0);           // c3 (dense)
    wth(S7, S8, 256, 32);                                             // w4 (1024 @16, padded)
    zb(S8, 16, 16, 1024);
    conv(S8, w4s, conv4_b, nullptr, T, 1024, 1024, 16, 9216, 0, 0);
    gnorm(T, S4, gn4_w, gn4_b, 1024, 256, 128, 10, 8, 4, 1);          // c4 (padded)

    // ---------------- Bottleneck ----------------
    zb(S4, 16, 16, 1024);
    conv(S4, w4s, conv4_b, nullptr, T, 1024, 1024, 16, 9216, 0, 0);
    gnorm(T, S6, gn4_w, gn4_b, 1024, 256, 128, 10, 8, 4, 1);          // c5 (padded)
    zb(S6, 16, 16, 1024);
    conv(S6, w4s, conv4_b, /*skip=*/S8, S4, 1024, 1024, 16, 9216, 1, 1);  // ic4 (dense f16)

    // ---------------- Decoder ----------------
    catcp(S7, S9, 256, 512, 0, 5, 10);                                // iw4[:, :256] = c3
    iwth(S4, S9, 256, 16, 512, 256, 1);                               // iw4[:, 256:]
    zb(S9, 32, 32, 512);
    conv(S9, d4s, convd4_b, nullptr, T, 512, 256, 32, 4608, 0, 0);
    gnorm(T, S8, gn3_w, gn3_b, 256, 1024, 32, 8, 10, 5, 0);           // ic3 (dense)

    catcp(S5, S9, 64, 128, 0, 6, 12);                                 // iw3[:, :64] = c2
    iwth(S8, S9, 64, 32, 128, 64, 1);
    zb(S9, 64, 64, 128);
    conv(S9, d3s, convd3_b, nullptr, T, 128, 64, 64, 1152, 0, 0);
    gnorm(T, S7, gn2_w, gn2_b, 64, 4096, 8, 6, 12, 6, 0);             // ic2 (dense)

    catcp(S3, S9, 16, 32, 0, 7, 14);                                  // iw2[:, :16] = c1
    iwth(S7, S9, 16, 64, 32, 16, 1);
    zb(S9, 128, 128, 32);
    conv(S9, d2s, convd2_b, nullptr, T, 32, 16, 128, 288, 0, 0);
    gnorm(T, S8, gn1_w, gn1_b, 16, 16384, 2, 4, 14, 7, 1);            // ic1 (padded)

    zb(S8, 128, 128, 16);
    conv(S8, d1s, convd1_b, nullptr, S2, 16, 12, 128, 160, 1, 1);     // iw1 (dense f16)
    iwth(S2, S6, 3, 128, 3, 0, 0);                                    // ixw (dense)

    {
        long total = (long)N * 2 * 65536;
        k_final_conv<<<ceil_div(total, 256), 256, 0, stream>>>(S6, fws, y, N, 65536);
    }
}
